// cRNN_17386027615004
// MI455X (gfx1250) — matrix operand order repaired and checked by reading, not compiled, not run on hardware
//
#include <hip/hip_runtime.h>

typedef _Float16 f16;
typedef __attribute__((ext_vector_type(16))) _Float16 v16h;
typedef __attribute__((ext_vector_type(8)))  _Float16 v8h;
typedef __attribute__((ext_vector_type(8)))  float    v8f;

#define B_   256
#define T_   168
#define LAG_ 168
#define LAT_ 128
#define WS_  64
#define H_   1024

// ---------- helpers ----------

// A-fragment (16x32 f16, M x K) per CDNA5 ISA layout:
// lanes 0-15: M=lane, K = k0..k0+7 (v0-3) and k0+16..k0+23 (v4-7)
// lanes 16-31: M=lane-16, K = k0+8..k0+15 and k0+24..k0+31
// caller passes base = row*ld + k0 + (lane>=16 ? 8 : 0)
__device__ __forceinline__ v16h load_a_frag(const f16* base) {
    v8h lo = *(const v8h*)(base);        // 8 consecutive f16 (16B)
    v8h hi = *(const v8h*)(base + 16);   // next K block (+16 elements)
    return __builtin_shufflevector(lo, hi, 0,1,2,3,4,5,6,7,8,9,10,11,12,13,14,15);
}

__device__ __forceinline__ float sigmoidf_(float x) {
    return 1.0f / (1.0f + __expf(-x));
}

// ---------- utility kernels ----------

__global__ void f32_to_f16_kernel(const float* __restrict__ src, f16* __restrict__ dst, int n) {
    int i = blockIdx.x * blockDim.x + threadIdx.x;
    if (i < n) dst[i] = (f16)src[i];
}

__global__ void init_out_kernel(float* __restrict__ out, const float* __restrict__ bo, int n) {
    int i = blockIdx.x * blockDim.x + threadIdx.x;
    if (i < n) out[i] = bo[0];
}

// ---------- encoder (tiny, fp32 scalar) ----------

__global__ void enc_layer_kernel(const float* __restrict__ in, const float* __restrict__ W,
                                 const float* __restrict__ b, float* __restrict__ out,
                                 int K, int N, int leaky) {
    int row = blockIdx.x;
    int j   = threadIdx.x;          // j < N
    const float* x = in + (size_t)row * K;
    const float* w = W  + (size_t)j   * K;
    float s = b[j];
    for (int k = 0; k < K; ++k) s += x[k] * w[k];
    if (leaky && s < 0.0f) s *= 0.01f;
    out[(size_t)row * N + j] = s;
}

__global__ void enc3_kernel(const float* __restrict__ in, const float* __restrict__ W,
                            const float* __restrict__ b,
                            float* __restrict__ h32, f16* __restrict__ h16) {
    int row = blockIdx.x;           // batch
    int j   = threadIdx.x;          // 0..1023
    const float* x = in + (size_t)row * LAT_;
    const float* w = W  + (size_t)j   * LAT_;
    float s = b[j];
    for (int k = 0; k < LAT_; ++k) s += x[k] * w[k];
    h32[(size_t)row * H_ + j] = s;
    h16[(size_t)row * H_ + j] = (f16)s;
}

// ---------- fused GRU step (WMMA) ----------
// grid = (16 ntiles of 64 cols, 16 btiles of 16 rows), block = 128 (4 waves).
// Each wave: one 16x16 output tile per gate; 4 f32 accumulators
// (r,z merged gi+gh; n kept split since n = tanh(in + r*hn)).

__global__ __launch_bounds__(128)
void gru_step_kernel(int t,
                     const float* __restrict__ hp32, const f16* __restrict__ hp16,
                     float* __restrict__ hn32,       f16* __restrict__ hn16,
                     const f16* __restrict__ Whh,    const f16* __restrict__ Wih,
                     const f16* __restrict__ X,      // [T,B,WS] f16
                     const float* __restrict__ bih,  const float* __restrict__ bhh,
                     const float* __restrict__ Wo,   float* __restrict__ out) {
    const int lane = threadIdx.x & 31;
    const int wv   = threadIdx.x >> 5;       // 0..3
    const int hl   = lane >> 4;              // half-wave: 0 or 1
    const int col  = lane & 15;              // N within tile / lane id
    const int nb   = blockIdx.x * 64 + wv * 16;  // output column base (0..1008)
    const int bt   = blockIdx.y * 16;            // batch row base

    v8f accR = {}, accZ = {}, accIN = {}, accHN = {};

    // ---- recurrent GEMM: gh = h_prev @ Whh^T, K = 1024 ----
    const f16* arow = hp16 + (size_t)(bt + col) * H_ + hl * 8;
    const f16* wR   = Whh  + (size_t)(nb + col) * H_ + hl * 8;
    const f16* wZ   = wR + (size_t)H_ * H_;
    const f16* wN   = wZ + (size_t)H_ * H_;
    for (int k0 = 0; k0 < H_; k0 += 32) {
        v16h a  = load_a_frag(arow + k0);
        v16h bR = load_a_frag(wR + k0);
        v16h bZ = load_a_frag(wZ + k0);
        v16h bN = load_a_frag(wN + k0);
        accR  = __builtin_amdgcn_wmma_f32_16x16x32_f16(false, a, false, bR, (short)0, accR,  false, false);
        accZ  = __builtin_amdgcn_wmma_f32_16x16x32_f16(false, a, false, bZ, (short)0, accZ,  false, false);
        accHN = __builtin_amdgcn_wmma_f32_16x16x32_f16(false, a, false, bN, (short)0, accHN, false, false);
    }

    // ---- input GEMM: gi = x_t @ Wih^T, K = 64 ----
    const f16* xrow = X   + ((size_t)t * B_ + bt + col) * WS_ + hl * 8;
    const f16* uR   = Wih + (size_t)(nb + col) * WS_ + hl * 8;
    const f16* uZ   = uR + (size_t)H_ * WS_;
    const f16* uN   = uZ + (size_t)H_ * WS_;
    for (int k0 = 0; k0 < WS_; k0 += 32) {
        v16h a  = load_a_frag(xrow + k0);
        v16h bR = load_a_frag(uR + k0);
        v16h bZ = load_a_frag(uZ + k0);
        v16h bN = load_a_frag(uN + k0);
        accR  = __builtin_amdgcn_wmma_f32_16x16x32_f16(false, a, false, bR, (short)0, accR,  false, false);
        accZ  = __builtin_amdgcn_wmma_f32_16x16x32_f16(false, a, false, bZ, (short)0, accZ,  false, false);
        accIN = __builtin_amdgcn_wmma_f32_16x16x32_f16(false, a, false, bN, (short)0, accIN, false, false);
    }

    // ---- gate math + state update + fused output projection ----
    const int   ncol  = nb + col;
    const float biasR = bih[ncol]        + bhh[ncol];
    const float biasZ = bih[H_ + ncol]   + bhh[H_ + ncol];
    const float bIN   = bih[2 * H_ + ncol];
    const float bHN   = bhh[2 * H_ + ncol];
    const float wo    = Wo[ncol];

    #pragma unroll
    for (int i = 0; i < 8; ++i) {
        // C/D layout: VGPR i -> M = i + hl*8, N = lane&15
        const int brow = bt + i + hl * 8;
        const float r  = sigmoidf_(accR[i] + biasR);
        const float z  = sigmoidf_(accZ[i] + biasZ);
        const float n  = tanhf(accIN[i] + bIN + r * (accHN[i] + bHN));
        const float hp = hp32[(size_t)brow * H_ + ncol];
        const float h  = (1.0f - z) * n + z * hp;
        hn32[(size_t)brow * H_ + ncol] = h;
        hn16[(size_t)brow * H_ + ncol] = (f16)h;

        // pred[b][t] += h * Wo[ncol]; lanes 0-15 (and 16-31) share the same brow
        float v = h * wo;
        v += __shfl_xor(v, 8, 16);
        v += __shfl_xor(v, 4, 16);
        v += __shfl_xor(v, 2, 16);
        v += __shfl_xor(v, 1, 16);
        if (col == 0) atomicAdd(&out[(size_t)brow * T_ + t], v);
    }
}

// ---------- host-side launch ----------

extern "C" void kernel_launch(void* const* d_in, const int* in_sizes, int n_in,
                              void* d_out, int out_size, void* d_ws, size_t ws_size,
                              hipStream_t stream) {
    (void)in_sizes; (void)n_in; (void)out_size; (void)ws_size;

    const float* lag  = (const float*)d_in[0];
    const float* curr = (const float*)d_in[1];
    const float* W1   = (const float*)d_in[2];
    const float* b1   = (const float*)d_in[3];
    const float* W2   = (const float*)d_in[4];
    const float* b2   = (const float*)d_in[5];
    const float* W3   = (const float*)d_in[6];
    const float* b3   = (const float*)d_in[7];
    const float* Wih  = (const float*)d_in[8];
    const float* Whh  = (const float*)d_in[9];
    const float* bih  = (const float*)d_in[10];
    const float* bhh  = (const float*)d_in[11];
    const float* Wo   = (const float*)d_in[12];
    const float* bo   = (const float*)d_in[13];
    float* out = (float*)d_out;

    // workspace carve-out (~15 MB)
    char* ws = (char*)d_ws;
    auto carve = [&](size_t bytes) -> char* {
        char* p = ws;
        ws += (bytes + 255) & ~(size_t)255;
        return p;
    };
    f16*   Wih_h = (f16*)carve((size_t)3 * H_ * WS_ * 2);
    f16*   Whh_h = (f16*)carve((size_t)3 * H_ * H_  * 2);
    f16*   X_h   = (f16*)carve((size_t)T_ * B_ * WS_ * 2);
    float* e1    = (float*)carve((size_t)B_ * 64   * 4);
    float* e2    = (float*)carve((size_t)B_ * LAT_ * 4);
    float* hA32  = (float*)carve((size_t)B_ * H_ * 4);
    float* hB32  = (float*)carve((size_t)B_ * H_ * 4);
    f16*   hA16  = (f16*)carve((size_t)B_ * H_ * 2);
    f16*   hB16  = (f16*)carve((size_t)B_ * H_ * 2);

    // convert weights / inputs to f16 once per call
    {
        int n;
        n = 3 * H_ * WS_;
        f32_to_f16_kernel<<<(n + 255) / 256, 256, 0, stream>>>(Wih, Wih_h, n);
        n = 3 * H_ * H_;
        f32_to_f16_kernel<<<(n + 255) / 256, 256, 0, stream>>>(Whh, Whh_h, n);
        n = T_ * B_ * WS_;
        f32_to_f16_kernel<<<(n + 255) / 256, 256, 0, stream>>>(curr, X_h, n);
    }

    // out[b][t] starts at bo (atomic partial sums accumulate onto it)
    init_out_kernel<<<(B_ * T_ + 255) / 256, 256, 0, stream>>>(out, bo, B_ * T_);

    // encoder MLP -> h0 (fp32 path, negligible cost)
    enc_layer_kernel<<<B_, 64,   0, stream>>>(lag, W1, b1, e1, LAG_, 64,   1);
    enc_layer_kernel<<<B_, LAT_, 0, stream>>>(e1,  W2, b2, e2, 64,   LAT_, 1);
    enc3_kernel<<<B_, H_, 0, stream>>>(e2, W3, b3, hA32, hA16);

    // 168 sequential fused GRU steps (double-buffered hidden state)
    for (int t = 0; t < T_; ++t) {
        const float* hp32 = (t & 1) ? hB32 : hA32;
        const f16*   hp16 = (t & 1) ? hB16 : hA16;
        float*       hn32 = (t & 1) ? hA32 : hB32;
        f16*         hn16 = (t & 1) ? hA16 : hB16;
        gru_step_kernel<<<dim3(16, 16), 128, 0, stream>>>(
            t, hp32, hp16, hn32, hn16, Whh_h, Wih_h, X_h, bih, bhh, Wo, out);
    }
}
